// StyleLinearLayer_61400852464041
// MI455X (gfx1250) — matrix operand order repaired and checked by reading, not compiled, not run on hardware
//
#include <hip/hip_runtime.h>
#include <hip/hip_bf16.h>

// ---------------------------------------------------------------------------
// StyleLinearLayer for MI455X (gfx1250).
//   out = (input @ weight.T) * (scale*|scale|*sqrt(rowsq(weight))) + bias
//   scale = style @ aff_w[:OUT].T + aff_b[:OUT]
// Fast path: fp32->bf16 pre-convert pass, then GEMM with a 3-deep
//   global_load_async_to_lds_b128 pipeline (ASYNCcnt) + ds_load_b128 + WMMA.
// Fallback (small ws): in-loop conversion via v_perm_b32 packing.
// ---------------------------------------------------------------------------

typedef __attribute__((ext_vector_type(8)))  __bf16 bfx8;
typedef __attribute__((ext_vector_type(16))) __bf16 bfx16;
typedef __attribute__((ext_vector_type(8)))  float  f32x8;

static constexpr int Bsz = 8192;
static constexpr int IN  = 2048;
static constexpr int OUT = 2048;
static constexpr int STY = 512;

static constexpr int BM = 128;
static constexpr int BN = 128;
static constexpr int BK = 32;
static constexpr int LDF = 40;   // LDS row stride, fallback kernel (elements)
static constexpr int LDT = 56;   // LDS row stride, async kernel (112B: 16B-aligned,
                                 // banks 28n mod 64 distinct for n=0..15)
static constexpr unsigned TILE_BYTES = BM * LDT * 2;  // one LDS tile buffer

// pack two fp32 -> packed bf16x2, round-to-nearest (ties away): add 0x8000,
// then v_perm_b32 picks the two high bytes of each source. 3 VALU total.
__device__ __forceinline__ unsigned int pack2_bf16(float a, float b) {
    unsigned int ua = __float_as_uint(a) + 0x8000u;
    unsigned int ub = __float_as_uint(b) + 0x8000u;
    return __builtin_amdgcn_perm(ub, ua, 0x07060302u);
}

// Assemble a 16-element bf16 fragment from two 8-element (16B) LDS chunks.
__device__ __forceinline__ bfx16 frag16(const __bf16* p0, const __bf16* p1) {
    bfx8 lo = *(const bfx8*)p0;
    bfx8 hi = *(const bfx8*)p1;
    return __builtin_shufflevector(lo, hi, 0, 1, 2, 3, 4, 5, 6, 7,
                                   8, 9, 10, 11, 12, 13, 14, 15);
}

// --------------------------- rsq = sqrt(rowsq) ------------------------------
__global__ __launch_bounds__(256) void rowsq_kernel(const float* __restrict__ w,
                                                    float* __restrict__ rsq) {
    const int row = blockIdx.x;
    const float4* p4 = (const float4*)(w + (size_t)row * IN);
    float4 a = p4[threadIdx.x];
    float4 b = p4[threadIdx.x + 256];
    float s = a.x * a.x + a.y * a.y + a.z * a.z + a.w * a.w +
              b.x * b.x + b.y * b.y + b.z * b.z + b.w * b.w;
    __shared__ float red[256];
    red[threadIdx.x] = s;
    __syncthreads();
#pragma unroll
    for (int off = 128; off > 0; off >>= 1) {
        if (threadIdx.x < off) red[threadIdx.x] += red[threadIdx.x + off];
        __syncthreads();
    }
    if (threadIdx.x == 0) rsq[row] = sqrtf(red[0]);
}

// ------------------------- fp32 -> bf16 pre-convert -------------------------
__global__ __launch_bounds__(256) void cvt_bf16_kernel(
    const float* __restrict__ src, __bf16* __restrict__ dst, long n4) {
    long i = (long)blockIdx.x * 256 + threadIdx.x;
    const long stride = (long)gridDim.x * 256;
    for (; i < n4; i += stride) {
        float4 v = ((const float4*)src)[i];
        uint2 u;
        u.x = pack2_bf16(v.x, v.y);
        u.y = pack2_bf16(v.z, v.w);
        ((uint2*)dst)[i] = u;
    }
}

// ---------------------- shared epilogue for both GEMMs ----------------------
__device__ __forceinline__ void epilogue(
    const f32x8 (&c)[4][2], const f32x8 (&s)[4][2],
    const float* __restrict__ bias, const float* __restrict__ aff_b,
    const float* __restrict__ rsq, float* __restrict__ out,
    int tileM, int tileN, int wm, int wn, int ln, int half) {
    // C/D layout: VGPR j, lane L -> M = j + (L>=16 ? 8 : 0), N = L&15.
#pragma unroll
    for (int ni = 0; ni < 2; ++ni) {
        const int n  = tileN + wn + ni * 16 + ln;
        const float rs = rsq[n];
        const float bb = bias[n];
        const float ab = aff_b[n];
#pragma unroll
        for (int mi = 0; mi < 4; ++mi) {
            const int mBase = tileM + wm + mi * 16 + half * 8;
#pragma unroll
            for (int j = 0; j < 8; ++j) {
                const float sc = s[mi][ni][j] + ab;
                const float mod = sc * fabsf(sc) * rs;
                out[(size_t)(mBase + j) * OUT + n] = c[mi][ni][j] * mod + bb;
            }
        }
    }
}

// ===================== FAST PATH: async bf16 GEMM ===========================
__global__ __launch_bounds__(256) void style_linear_async_kernel(
    const __bf16* __restrict__ inputB, const __bf16* __restrict__ styleB,
    const __bf16* __restrict__ weightB, const __bf16* __restrict__ affwB,
    const float* __restrict__ bias, const float* __restrict__ aff_b,
    const float* __restrict__ rsq, float* __restrict__ out) {

    __shared__ __bf16 As[3][BM * LDT];   // 3-deep ring (issue depth 2)
    __shared__ __bf16 Bs[3][BN * LDT];

    const int tileM = blockIdx.y * BM;
    const int tileN = blockIdx.x * BN;
    const int wave  = threadIdx.x >> 5;
    const int lane  = threadIdx.x & 31;
    const int wm    = (wave & 1) * 64;
    const int wn    = (wave >> 1) * 32;
    const int ln    = lane & 15;
    const int half  = lane >> 4;
    const int aBase = half * 8;   // A frag: K chunks {aBase..+7, aBase+16..+23}
    const int bBase = half * 8;   // B frag: K chunks {bBase..+7, bBase+16..+23}

    // Per-lane copy-chunk coords: tile = 512 x 16B chunks, 256 lanes x 2.
    const int r0  = threadIdx.x >> 2;          // rows 0..63
    const int c0e = (threadIdx.x & 3) * 8;     // element col: 0,8,16,24
    const int r1  = r0 + 64;                   // rows 64..127 (chunk 1)

    // Loop-invariant LDS byte addresses (low 32 bits of flat ptr == LDS addr).
    const unsigned ldsA0 = (unsigned)(size_t)(const void*)&As[0][r0 * LDT + c0e];
    const unsigned ldsA1 = (unsigned)(size_t)(const void*)&As[0][r1 * LDT + c0e];
    const unsigned ldsB0 = (unsigned)(size_t)(const void*)&Bs[0][r0 * LDT + c0e];
    const unsigned ldsB1 = (unsigned)(size_t)(const void*)&Bs[0][r1 * LDT + c0e];

    f32x8 c[4][2];
    f32x8 s[4][2];
    const f32x8 zero = {0.f, 0.f, 0.f, 0.f, 0.f, 0.f, 0.f, 0.f};
#pragma unroll
    for (int mi = 0; mi < 4; ++mi)
#pragma unroll
        for (int ni = 0; ni < 2; ++ni) { c[mi][ni] = zero; s[mi][ni] = zero; }

    // Issue one tile's worth for this lane (2 x 16B) and bump the pointers.
    auto issue2 = [&](const char*& p0, const char*& p1,
                      unsigned l0, unsigned l1) {
        asm volatile("global_load_async_to_lds_b128 %0, %1, off"
                     :: "v"(l0), "v"(p0) : "memory");
        asm volatile("global_load_async_to_lds_b128 %0, %1, off"
                     :: "v"(l1), "v"(p1) : "memory");
        p0 += BK * 2;   // advance one k-step (bytes)
        p1 += BK * 2;
    };

    auto compute_step = [&](const __bf16* Ab, const __bf16* Bb,
                            f32x8 (&acc)[4][2]) {
        bfx16 bf[2];
#pragma unroll
        for (int ni = 0; ni < 2; ++ni) {
            const __bf16* p = &Bb[(wn + ni * 16 + ln) * LDT + bBase];
            bf[ni] = frag16(p, p + 16);
        }
#pragma unroll
        for (int mi = 0; mi < 4; ++mi) {
            const __bf16* p = &Ab[(wm + mi * 16 + ln) * LDT + aBase];
            bfx16 af = frag16(p, p + 16);
#pragma unroll
            for (int ni = 0; ni < 2; ++ni)
                acc[mi][ni] = __builtin_amdgcn_wmma_f32_16x16x32_bf16(
                    false, af, false, bf[ni], (short)0, acc[mi][ni],
                    false, false);
        }
    };

    auto run_phase = [&](const __bf16* A, const __bf16* Bm, int ldk, int K,
                         f32x8 (&acc)[4][2]) {
        // Per-lane running global byte pointers (strength-reduced).
        const char* pa0 = (const char*)(A + (size_t)(tileM + r0) * ldk + c0e);
        const char* pa1 = (const char*)(A + (size_t)(tileM + r1) * ldk + c0e);
        const char* pb0 = (const char*)(Bm + (size_t)(tileN + r0) * ldk + c0e);
        const char* pb1 = (const char*)(Bm + (size_t)(tileN + r1) * ldk + c0e);

        // Prologue: tile0 -> buf0, tile1 -> buf1 (8 async issues in flight).
        issue2(pa0, pa1, ldsA0, ldsA1);
        issue2(pb0, pb1, ldsB0, ldsB1);
        issue2(pa0, pa1, ldsA0 + TILE_BYTES, ldsA1 + TILE_BYTES);
        issue2(pb0, pb1, ldsB0 + TILE_BYTES, ldsB1 + TILE_BYTES);

        int buf = 0;
#pragma unroll 1
        for (int kk = 0; kk < K; kk += BK) {
            // In-order completion: <=4 outstanding => current tile's 4 copies
            // are done while the next tile's may still be in flight.
            if (kk + 2 * BK < K) {
                asm volatile("s_wait_asynccnt 4" ::: "memory");
            } else {
                asm volatile("s_wait_asynccnt 0" ::: "memory");
            }
            __syncthreads();  // buffer `buf` fully populated for all waves;
                              // everyone done reading the re-use target buffer
            if (kk + 2 * BK < K) {
                const int nb = (buf == 0) ? 2 : buf - 1;   // (buf+2)%3
                const unsigned nbo = (unsigned)nb * TILE_BYTES;
                issue2(pa0, pa1, ldsA0 + nbo, ldsA1 + nbo);
                issue2(pb0, pb1, ldsB0 + nbo, ldsB1 + nbo);
            }
            compute_step(&As[buf][0], &Bs[buf][0], acc);
            buf = (buf == 2) ? 0 : buf + 1;
        }
        __syncthreads();
    };

    run_phase(styleB, affwB, STY, STY, s);   // scale = style @ aff_w[:OUT].T
    run_phase(inputB, weightB, IN, IN, c);   // base  = input @ weight.T

    epilogue(c, s, bias, aff_b, rsq, out, tileM, tileN, wm, wn, ln, half);
}

// ================== FALLBACK: in-loop-conversion GEMM =======================
struct TileR { float4 v[4]; };

__device__ __forceinline__ TileR load_tile_regs(const float* __restrict__ g,
                                                int ldg, int row0, int col0) {
    TileR t;
    const int tid = threadIdx.x;
#pragma unroll
    for (int i = 0; i < 4; ++i) {
        const int p  = tid + i * 256;
        const int r  = p >> 3;
        const int c4 = p & 7;
        t.v[i] = *(const float4*)(g + (size_t)(row0 + r) * ldg + col0 + c4 * 4);
    }
    return t;
}

__device__ __forceinline__ void store_tile_lds(const TileR& t, __bf16* lds) {
    const int tid = threadIdx.x;
#pragma unroll
    for (int i = 0; i < 4; ++i) {
        const int p  = tid + i * 256;
        const int r  = p >> 3;
        const int c4 = p & 7;
        uint2 u;
        u.x = pack2_bf16(t.v[i].x, t.v[i].y);
        u.y = pack2_bf16(t.v[i].z, t.v[i].w);
        *(uint2*)&lds[r * LDF + c4 * 4] = u;
    }
}

__global__ __launch_bounds__(256) void style_linear_fallback_kernel(
    const float* __restrict__ input, const float* __restrict__ style,
    const float* __restrict__ weight, const float* __restrict__ bias,
    const float* __restrict__ aff_w, const float* __restrict__ aff_b,
    const float* __restrict__ rsq, float* __restrict__ out) {

    __shared__ __bf16 As[BM * LDF];
    __shared__ __bf16 Bs[BN * LDF];

    const int tileM = blockIdx.y * BM;
    const int tileN = blockIdx.x * BN;
    const int wave  = threadIdx.x >> 5;
    const int lane  = threadIdx.x & 31;
    const int wm    = (wave & 1) * 64;
    const int wn    = (wave >> 1) * 32;
    const int ln    = lane & 15;
    const int half  = lane >> 4;
    const int aBase = half * 8;
    const int bBase = half * 8;

    f32x8 c[4][2];
    f32x8 s[4][2];
    const f32x8 zero = {0.f, 0.f, 0.f, 0.f, 0.f, 0.f, 0.f, 0.f};
#pragma unroll
    for (int mi = 0; mi < 4; ++mi)
#pragma unroll
        for (int ni = 0; ni < 2; ++ni) { c[mi][ni] = zero; s[mi][ni] = zero; }

    auto compute_step = [&](f32x8 (&acc)[4][2]) {
        bfx16 bf[2];
#pragma unroll
        for (int ni = 0; ni < 2; ++ni) {
            const __bf16* p = &Bs[(wn + ni * 16 + ln) * LDF + bBase];
            bf[ni] = frag16(p, p + 16);
        }
#pragma unroll
        for (int mi = 0; mi < 4; ++mi) {
            const __bf16* p = &As[(wm + mi * 16 + ln) * LDF + aBase];
            bfx16 af = frag16(p, p + 16);
#pragma unroll
            for (int ni = 0; ni < 2; ++ni)
                acc[mi][ni] = __builtin_amdgcn_wmma_f32_16x16x32_bf16(
                    false, af, false, bf[ni], (short)0, acc[mi][ni],
                    false, false);
        }
    };

    TileR ra = load_tile_regs(style, STY, tileM, 0);
    TileR rb = load_tile_regs(aff_w, STY, tileN, 0);
#pragma unroll 1
    for (int kk = 0; kk < STY; kk += BK) {
        store_tile_lds(ra, As);
        store_tile_lds(rb, Bs);
        __syncthreads();
        if (kk + BK < STY) {
            ra = load_tile_regs(style, STY, tileM, kk + BK);
            rb = load_tile_regs(aff_w, STY, tileN, kk + BK);
        } else {
            ra = load_tile_regs(input, IN, tileM, 0);
            rb = load_tile_regs(weight, IN, tileN, 0);
        }
        compute_step(s);
        __syncthreads();
    }
#pragma unroll 1
    for (int kk = 0; kk < IN; kk += BK) {
        store_tile_lds(ra, As);
        store_tile_lds(rb, Bs);
        __syncthreads();
        if (kk + BK < IN) {
            ra = load_tile_regs(input, IN, tileM, kk + BK);
            rb = load_tile_regs(weight, IN, tileN, kk + BK);
        }
        compute_step(c);
        __syncthreads();
    }

    epilogue(c, s, bias, aff_b, rsq, out, tileM, tileN, wm, wn, ln, half);
}

// ---------------------------------------------------------------------------
extern "C" void kernel_launch(void* const* d_in, const int* in_sizes, int n_in,
                              void* d_out, int out_size, void* d_ws,
                              size_t ws_size, hipStream_t stream) {
    const float* input  = (const float*)d_in[0];
    const float* style  = (const float*)d_in[1];
    const float* weight = (const float*)d_in[2];
    const float* bias   = (const float*)d_in[3];
    const float* aff_w  = (const float*)d_in[4];
    const float* aff_b  = (const float*)d_in[5];
    float* out = (float*)d_out;

    float* rsq = (float*)d_ws;             // 8 KB at the front of ws
    const size_t base = 8192;
    const size_t nIn  = (size_t)Bsz * IN;
    const size_t nSty = (size_t)Bsz * STY;
    const size_t nW   = (size_t)OUT * IN;
    const size_t nAff = (size_t)OUT * STY; // only first OUT rows of aff_w used
    const size_t need = base + 2 * (nIn + nSty + nW + nAff);

    rowsq_kernel<<<OUT, 256, 0, stream>>>(weight, rsq);

    dim3 grid(OUT / BN, Bsz / BM);   // (16, 64)
    if (ws_size >= need) {
        __bf16* inB  = (__bf16*)((char*)d_ws + base);
        __bf16* styB = inB + nIn;
        __bf16* wB   = styB + nSty;
        __bf16* affB = wB + nW;
        cvt_bf16_kernel<<<2048, 256, 0, stream>>>(input, inB, (long)(nIn / 4));
        cvt_bf16_kernel<<<1024, 256, 0, stream>>>(style, styB, (long)(nSty / 4));
        cvt_bf16_kernel<<<1024, 256, 0, stream>>>(weight, wB, (long)(nW / 4));
        cvt_bf16_kernel<<<512, 256, 0, stream>>>(aff_w, affB, (long)(nAff / 4));
        style_linear_async_kernel<<<grid, 256, 0, stream>>>(
            inB, styB, wB, affB, bias, aff_b, rsq, out);
    } else {
        style_linear_fallback_kernel<<<grid, 256, 0, stream>>>(
            input, style, weight, bias, aff_w, aff_b, rsq, out);
    }
}
